// GraphEncoder_48103633715564
// MI455X (gfx1250) — matrix operand order repaired and checked by reading, not compiled, not run on hardware
//
#include <hip/hip_runtime.h>

// ---------------------------------------------------------------------------
// Problem constants (from the reference)
// ---------------------------------------------------------------------------
static constexpr int BS  = 2048;
static constexpr int NN  = 100000;   // nodes
static constexpr int EE_ = 400000;   // edges
static constexpr int SS  = 8192;     // shapes
static constexpr int ESH = 12288;    // hyper edges
static constexpr int LNS = 100000;   // node-in-shape entries
static constexpr int NL  = 20000;    // leafs
static constexpr int DD  = 256;
static constexpr int EH  = 64;
static constexpr int GF  = 64;
static constexpr int D2  = 512;

typedef __attribute__((ext_vector_type(16))) __bf16         bf16x16;
typedef __attribute__((ext_vector_type(8)))  float          floatx8;
typedef __attribute__((ext_vector_type(4)))  unsigned int   uintx4;
typedef __attribute__((ext_vector_type(8)))  unsigned short ushortx8;

__device__ __forceinline__ unsigned short f2bf(float f) {
  unsigned int u = __float_as_uint(f);
  unsigned int r = u + 0x7FFFu + ((u >> 16) & 1u);   // round-to-nearest-even
  return (unsigned short)(r >> 16);
}
__device__ __forceinline__ float bf2f(unsigned short h) {
  return __uint_as_float(((unsigned int)h) << 16);
}

// ---------------------------------------------------------------------------
// Elementwise helpers
// ---------------------------------------------------------------------------
__global__ void k_cast_f32_bf16(const float* __restrict__ src,
                                unsigned short* __restrict__ dst, size_t n) {
  size_t i = (size_t)blockIdx.x * blockDim.x + threadIdx.x;
  size_t stride = (size_t)gridDim.x * blockDim.x;
  for (; i < n; i += stride) dst[i] = f2bf(src[i]);
}

__global__ void k_zero_f32(float* __restrict__ p, size_t n) {
  size_t i = (size_t)blockIdx.x * blockDim.x + threadIdx.x;
  size_t stride = (size_t)gridDim.x * blockDim.x;
  for (; i < n; i += stride) p[i] = 0.0f;
}

// eattr gather: out[e, :] = bond_table[bond_types[e], :]  (bf16)
__global__ void k_gather_bonds(const float* __restrict__ bond_table,
                               const int* __restrict__ bond_types,
                               unsigned short* __restrict__ out, int ne, int eh) {
  int e = blockIdx.x;
  if (e >= ne) return;
  const float* row = bond_table + (size_t)bond_types[e] * eh;
  unsigned short* op = out + (size_t)e * eh;
  for (int c = threadIdx.x; c < eh; c += blockDim.x) op[c] = f2bf(row[c]);
}

// ---------------------------------------------------------------------------
// WMMA bf16 GEMM:  out = X[M,K] @ W[K,Nc] + bias, optional relu.
// Tile: 128x64 per block (256 threads = 8 waves). K staged in 32-chunks with
// LDS double-buffering; fragments use contiguous k = 16*half + s (A and B alike) and are
// fetched with ds_load_b128 (LDS row stride 40 halfwords keeps 16B alignment).
// ---------------------------------------------------------------------------
#define TM 128
#define TN 64
#define TK 32
#define LDSTR 40                 // halfwords per LDS row (80B, 16B aligned)
#define GEMM_THREADS 256

__global__ __launch_bounds__(GEMM_THREADS)
void k_gemm_bf16_wmma(const unsigned short* __restrict__ X,   // [M,K] bf16
                      const unsigned short* __restrict__ W,   // [K,Nc] bf16
                      const float* __restrict__ bias,          // [Nc] or null
                      float* __restrict__ outF,                // [M,Nc] or null
                      unsigned short* __restrict__ outB,       // [M,Nc] or null
                      int M, int K, int Nc, int relu) {
  __shared__ unsigned short As[2][TM * LDSTR];  // [row][k]
  __shared__ unsigned short Bs[2][TN * LDSTR];  // transposed: [n][k]

  const int tid  = threadIdx.x;
  const int wave = tid >> 5;
  const int lane = tid & 31;
  const int half = lane >> 4;
  const int l16  = lane & 15;

  const int m0 = blockIdx.x * TM;
  const int n0 = blockIdx.y * TN;

  floatx8 acc[4] = {};
  uintx4 ar[2];  // A staging regs: 2 x b128 per thread
  uintx4 br;     // B staging reg : 1 x b128 per thread

  // ---- staging helpers (fully unrolled, vector b128) ----------------------
  auto loadA = [&](int kk) {
#pragma unroll
    for (int u = 0; u < 2; ++u) {
      int idx = tid + u * GEMM_THREADS;      // 0..511
      int r = idx >> 2, c = idx & 3;         // row, 8-elem chunk
      int gr = m0 + r;
      uintx4 v = {};
      if (gr < M)
        v = *reinterpret_cast<const uintx4*>(X + (size_t)gr * K + kk + c * 8);
      ar[u] = v;
    }
  };
  auto storeA = [&](int buf) {
#pragma unroll
    for (int u = 0; u < 2; ++u) {
      int idx = tid + u * GEMM_THREADS;
      int r = idx >> 2, c = idx & 3;
      *reinterpret_cast<uintx4*>(&As[buf][r * LDSTR + c * 8]) = ar[u];
    }
  };
  auto loadB = [&](int kk) {
    int k = tid >> 3, c = tid & 7;           // 32 k-rows x 8 chunks
    br = *reinterpret_cast<const uintx4*>(W + (size_t)(kk + k) * Nc + n0 + c * 8);
  };
  auto storeB = [&](int buf) {
    int k = tid >> 3, c = tid & 7;
    ushortx8 h = __builtin_bit_cast(ushortx8, br);
#pragma unroll
    for (int j = 0; j < 8; ++j)
      Bs[buf][(c * 8 + j) * LDSTR + k] = h[j];
  };

  // ---- prologue: stage tile 0 --------------------------------------------
  loadA(0);
  loadB(0);
  storeA(0);
  storeB(0);
  __syncthreads();

  int cur = 0;
  for (int kk = 0; kk < K; kk += TK) {
    const bool have_next = (kk + TK) < K;
    if (have_next) {                     // overlap next-tile loads with WMMA
      loadA(kk + TK);
      loadB(kk + TK);
    }
    if (kk + 2 * TK < K && (m0 + (tid >> 1)) < M)
      __builtin_prefetch(X + (size_t)(m0 + (tid >> 1)) * K + kk + 2 * TK, 0, 0);

    // ---- A fragment: two contiguous 16B runs -> 2 x ds_load_b128 ---------
    union { uintx4 q[2]; bf16x16 v; } aa;
    {
      const unsigned short* ap = &As[cur][(wave * 16 + l16) * LDSTR + 16 * half];
      aa.q[0] = *reinterpret_cast<const uintx4*>(ap);
      aa.q[1] = *reinterpret_cast<const uintx4*>(ap + 8);
    }
#pragma unroll
    for (int t = 0; t < 4; ++t) {
      // B fragment: 16 contiguous k-halfwords -> 2 x ds_load_b128
      union { uintx4 q[2]; bf16x16 v; } bb;
      const unsigned short* bp = &Bs[cur][(t * 16 + l16) * LDSTR + 16 * half];
      bb.q[0] = *reinterpret_cast<const uintx4*>(bp);
      bb.q[1] = *reinterpret_cast<const uintx4*>(bp + 8);
      acc[t] = __builtin_amdgcn_wmma_f32_16x16x32_bf16(
          false, aa.v, false, bb.v, (short)0, acc[t], false, false);
    }

    if (have_next) {
      storeA(cur ^ 1);
      storeB(cur ^ 1);
      __syncthreads();
      cur ^= 1;
    }
  }

  // ---- write back: C element g -> row = 16*wave + g + 8*half, col = l16 ---
#pragma unroll
  for (int t = 0; t < 4; ++t) {
    int gc = n0 + t * 16 + l16;
    float b = bias ? bias[gc] : 0.0f;
#pragma unroll
    for (int g = 0; g < 8; ++g) {
      int gr = m0 + wave * 16 + g + 8 * half;
      if (gr < M) {
        float v = acc[t][g] + b;
        if (relu) v = fmaxf(v, 0.0f);
        if (outF) outF[(size_t)gr * Nc + gc] = v;
        if (outB) outB[(size_t)gr * Nc + gc] = f2bf(v);
      }
    }
  }
}

// ---------------------------------------------------------------------------
// TransformerConv edge machinery (softmax via atomics; denom divided later)
// ---------------------------------------------------------------------------
__global__ void k_edge_logits(const unsigned short* __restrict__ Q,
                              const unsigned short* __restrict__ Kc,
                              const unsigned short* __restrict__ Ea,
                              const int* __restrict__ src, const int* __restrict__ dst,
                              float* __restrict__ ex, float* __restrict__ den,
                              int ne, int C, float scale) {
  int e = blockIdx.x * (blockDim.x >> 5) + (threadIdx.x >> 5);
  if (e >= ne) return;
  int lane = threadIdx.x & 31;
  int s = src[e], d = dst[e];
  const unsigned short* qp = Q + (size_t)d * C;
  const unsigned short* kp = Kc + (size_t)s * C;
  const unsigned short* ep = Ea + (size_t)e * C;
  float acc = 0.f;
  for (int c = lane; c < C; c += 32)
    acc += bf2f(qp[c]) * (bf2f(kp[c]) + bf2f(ep[c]));
  for (int off = 16; off > 0; off >>= 1)
    acc += __shfl_down(acc, off, 32);
  if (lane == 0) {
    float v = __expf(acc * scale);
    ex[e] = v;
    atomicAdd(&den[d], v);
  }
}

__global__ void k_edge_scatter(const unsigned short* __restrict__ V,
                               const unsigned short* __restrict__ Ea,
                               const int* __restrict__ src, const int* __restrict__ dst,
                               const float* __restrict__ ex,
                               float* __restrict__ num, int ne, int C) {
  int e = blockIdx.x;
  if (e >= ne) return;
  int s = src[e], d = dst[e];
  float a = ex[e];
  const unsigned short* vp = V + (size_t)s * C;
  const unsigned short* ep = Ea + (size_t)e * C;
  float* op = num + (size_t)d * C;
  for (int c = threadIdx.x; c < C; c += blockDim.x)
    atomicAdd(&op[c], (bf2f(vp[c]) + bf2f(ep[c])) * a);
}

__global__ void k_combine_relu(const float* __restrict__ num, const float* __restrict__ den,
                               const float* __restrict__ skip,
                               unsigned short* __restrict__ hout, int M, int C) {
  size_t i = (size_t)blockIdx.x * blockDim.x + threadIdx.x;
  size_t n = (size_t)M * C;
  if (i >= n) return;
  int r = (int)(i / C);
  float d = den[r];
  float a = (d > 0.f) ? num[i] / d : 0.f;
  hout[i] = f2bf(fmaxf(a + skip[i], 0.f));
}

// ---------------------------------------------------------------------------
// Segment aggregations (atomics; targets are small & L2-resident)
// ---------------------------------------------------------------------------
__global__ void k_accum_rows_bf16(const unsigned short* __restrict__ src,
                                  const int* __restrict__ seg,
                                  float* __restrict__ out, int n, int C) {
  int i = blockIdx.x;
  if (i >= n) return;
  float* op = out + (size_t)seg[i] * C;
  const unsigned short* sp = src + (size_t)i * C;
  for (int c = threadIdx.x; c < C; c += blockDim.x) atomicAdd(&op[c], bf2f(sp[c]));
}

// shape_emb[s] += concat(node_feats[a], h[a])  (reads bf16 copies)
__global__ void k_accum_ne_seg(const unsigned short* __restrict__ XB,
                               const unsigned short* __restrict__ HB,
                               const int* __restrict__ idx_atom, const int* __restrict__ idx_seg,
                               float* __restrict__ out, int n, int Dm) {
  int i = blockIdx.x;
  if (i >= n) return;
  int a = idx_atom[i];
  float* op = out + (size_t)idx_seg[i] * (2 * Dm);
  const unsigned short* xp = XB + (size_t)a * Dm;
  const unsigned short* hp = HB + (size_t)a * Dm;
  for (int c = threadIdx.x; c < Dm; c += blockDim.x) {
    atomicAdd(&op[c], bf2f(xp[c]));
    atomicAdd(&op[Dm + c], bf2f(hp[c]));
  }
}

// join_emb (bf16, for GEMM) + join_agg accumulation in one pass
__global__ void k_build_join(const float* __restrict__ she,
                             const unsigned short* __restrict__ XB,
                             const unsigned short* __restrict__ HB,
                             const int* __restrict__ hrow, const int* __restrict__ hcol,
                             const int* __restrict__ jidx, const int* __restrict__ jmol,
                             unsigned short* __restrict__ joinb, float* __restrict__ join_agg,
                             int nj, int Dm, int Dm2) {
  int j = blockIdx.x;
  if (j >= nj) return;
  const float* r0 = she + (size_t)hrow[j] * Dm2;
  const float* r1 = she + (size_t)hcol[j] * Dm2;
  int a = jidx[j];
  const unsigned short* xp = XB + (size_t)a * Dm;
  const unsigned short* hp = HB + (size_t)a * Dm;
  unsigned short* ob = joinb + (size_t)j * (3 * Dm2);
  float* oa = join_agg + (size_t)jmol[j] * (3 * Dm2);
  for (int c = threadIdx.x; c < Dm2; c += blockDim.x) {
    float v0 = r0[c], v1 = r1[c];
    float v2 = (c < Dm) ? bf2f(xp[c]) : bf2f(hp[c - Dm]);
    ob[c] = f2bf(v0);
    ob[Dm2 + c] = f2bf(v1);
    ob[2 * Dm2 + c] = f2bf(v2);
    atomicAdd(&oa[c], v0);
    atomicAdd(&oa[Dm2 + c], v1);
    atomicAdd(&oa[2 * Dm2 + c], v2);
  }
}

__global__ void k_build_leaf(const float* __restrict__ she,
                             const unsigned short* __restrict__ XB,
                             const unsigned short* __restrict__ HB,
                             const int* __restrict__ lnode, const int* __restrict__ lattach,
                             const int* __restrict__ lprim, const int* __restrict__ lsec,
                             const int* __restrict__ lmol,
                             float* __restrict__ leaf_agg, int nl, int Dm, int Dm2) {
  int j = blockIdx.x;
  if (j >= nl) return;
  int nidx = lnode[j], aidx = lattach[j], p = lprim[j], sc = lsec[j];
  float* oa = leaf_agg + (size_t)lmol[j] * (4 * Dm2);
  for (int c = threadIdx.x; c < Dm2; c += blockDim.x) {
    float ne_n = (c < Dm) ? bf2f(XB[(size_t)nidx * Dm + c]) : bf2f(HB[(size_t)nidx * Dm + c - Dm]);
    float ne_a = (c < Dm) ? bf2f(XB[(size_t)aidx * Dm + c]) : bf2f(HB[(size_t)aidx * Dm + c - Dm]);
    float pe = she[(size_t)p * Dm2 + c];
    float se = (sc >= 0) ? she[(size_t)sc * Dm2 + c] : 0.f;
    atomicAdd(&oa[c], ne_n);
    atomicAdd(&oa[Dm2 + c], ne_a);
    atomicAdd(&oa[2 * Dm2 + c], pe);
    atomicAdd(&oa[3 * Dm2 + c], se);
  }
}

// shape_agg[mol] += concat(shape_emb[s] (f32), sh_final[s] (bf16))
__global__ void k_accum_shape_cat(const float* __restrict__ she,
                                  const unsigned short* __restrict__ shb,
                                  const int* __restrict__ seg, float* __restrict__ out,
                                  int ns, int Dm2) {
  int s = blockIdx.x;
  if (s >= ns) return;
  float* op = out + (size_t)seg[s] * (2 * Dm2);
  for (int c = threadIdx.x; c < Dm2; c += blockDim.x) {
    atomicAdd(&op[c], she[(size_t)s * Dm2 + c]);
    atomicAdd(&op[Dm2 + c], bf2f(shb[(size_t)s * Dm2 + c]));
  }
}

// z = concat([glob0,node0,shape0,join0,leaf0, glob1,node1,shape1,join1,leaf1])
__global__ void k_assemble_z(const float* __restrict__ lg, const float* __restrict__ ln,
                             const float* __restrict__ ls, const float* __restrict__ lj,
                             const float* __restrict__ ll, float* __restrict__ z, int bs) {
  int b = blockIdx.x;
  if (b >= bs) return;
  for (int c = threadIdx.x; c < 1024; c += blockDim.x) {
    int h = (c >= 512) ? 1 : 0;
    int cc = c - 512 * h;
    float v;
    if (cc < 64)        v = lg[(size_t)b * 128 + h * 64  + cc];
    else if (cc < 192)  v = ln[(size_t)b * 256 + h * 128 + (cc - 64)];
    else if (cc < 320)  v = ls[(size_t)b * 256 + h * 128 + (cc - 192)];
    else if (cc < 416)  v = lj[(size_t)b * 192 + h * 96  + (cc - 320)];
    else                v = ll[(size_t)b * 192 + h * 96  + (cc - 416)];
    z[(size_t)b * 1024 + c] = v;
  }
}

// ---------------------------------------------------------------------------
// Host orchestration
// ---------------------------------------------------------------------------
extern "C" void kernel_launch(void* const* d_in, const int* in_sizes, int n_in,
                              void* d_out, int out_size, void* d_ws, size_t ws_size,
                              hipStream_t stream) {
  (void)in_sizes; (void)n_in; (void)out_size; (void)ws_size;

  // ---- inputs (setup_inputs dict order) ----
  const float* node_feats     = (const float*)d_in[0];
  const float* graph_features = (const float*)d_in[1];
  const float* bond_table     = (const float*)d_in[2];
  const float* aq_w = (const float*)d_in[3];  const float* aq_b = (const float*)d_in[4];
  const float* ak_w = (const float*)d_in[5];  const float* ak_b = (const float*)d_in[6];
  const float* av_w = (const float*)d_in[7];  const float* av_b = (const float*)d_in[8];
  const float* ae_w = (const float*)d_in[9];
  const float* as_w = (const float*)d_in[10]; const float* as_b = (const float*)d_in[11];
  const float* sq_w = (const float*)d_in[12]; const float* sq_b = (const float*)d_in[13];
  const float* sk_w = (const float*)d_in[14]; const float* sk_b = (const float*)d_in[15];
  const float* sv_w = (const float*)d_in[16]; const float* sv_b = (const float*)d_in[17];
  const float* se_w = (const float*)d_in[18];
  const float* ss_w = (const float*)d_in[19]; const float* ss_b = (const float*)d_in[20];
  const float* m_atoms_w  = (const float*)d_in[21]; const float* m_atoms_b  = (const float*)d_in[22];
  const float* m_shapes_w = (const float*)d_in[23]; const float* m_shapes_b = (const float*)d_in[24];
  const float* m_joins_w  = (const float*)d_in[25]; const float* m_joins_b  = (const float*)d_in[26];
  const float* m_leafs_w  = (const float*)d_in[27]; const float* m_leafs_b  = (const float*)d_in[28];
  const float* m_glob_w   = (const float*)d_in[29]; const float* m_glob_b   = (const float*)d_in[30];
  const int* edge_src   = (const int*)d_in[31];
  const int* edge_dst   = (const int*)d_in[32];
  const int* bond_types = (const int*)d_in[33];
  const int* node_mol_id = (const int*)d_in[34];
  const int* nis_atom = (const int*)d_in[35];
  const int* nis_shape = (const int*)d_in[36];
  const int* hyper_row = (const int*)d_in[37];
  const int* hyper_col = (const int*)d_in[38];
  const int* join_idxs = (const int*)d_in[39];
  const int* join_mol_id = (const int*)d_in[40];
  const int* shape_mol_id = (const int*)d_in[41];
  const int* leaf_node_idx = (const int*)d_in[42];
  const int* leaf_attach_idx = (const int*)d_in[43];
  const int* leaf_prim_shape = (const int*)d_in[44];
  const int* leaf_mol_id = (const int*)d_in[45];
  const int* leaf_sec_shape = (const int*)d_in[46];

  // ---- workspace layout ----
  char* base = (char*)d_ws;
  size_t off = 0;
  auto alloc = [&](size_t bytes) -> void* {
    void* p = base + off;
    off = (off + bytes + 255) & ~(size_t)255;
    return p;
  };
  typedef unsigned short bf;
  bf* XB   = (bf*)alloc((size_t)NN * DD * 2);
  bf* HB0  = (bf*)alloc((size_t)NN * DD * 2);
  bf* HB1  = (bf*)alloc((size_t)NN * DD * 2);
  bf* QB   = (bf*)alloc((size_t)NN * DD * 2);
  bf* KB   = (bf*)alloc((size_t)NN * DD * 2);
  bf* VB   = (bf*)alloc((size_t)NN * DD * 2);
  bf* EATTRB = (bf*)alloc((size_t)EE_ * EH * 2);
  bf* EEB  = (bf*)alloc((size_t)EE_ * DD * 2);
  float* NUM  = (float*)alloc((size_t)NN * DD * 4);
  float* SKIP = (float*)alloc((size_t)NN * DD * 4);
  float* DEN  = (float*)alloc((size_t)NN * 4);
  float* EX   = (float*)alloc((size_t)EE_ * 4);
  // phase B
  float* SHE  = (float*)alloc((size_t)SS * D2 * 4);
  bf* SHEB    = (bf*)alloc((size_t)SS * D2 * 2);
  bf* SHB0    = (bf*)alloc((size_t)SS * D2 * 2);
  bf* SHB1    = (bf*)alloc((size_t)SS * D2 * 2);
  bf* JOINB   = (bf*)alloc((size_t)ESH * 3 * D2 * 2);
  bf* QSB     = (bf*)alloc((size_t)SS * D2 * 2);
  bf* KSB     = (bf*)alloc((size_t)SS * D2 * 2);
  bf* VSB     = (bf*)alloc((size_t)SS * D2 * 2);
  bf* ESB     = (bf*)alloc((size_t)ESH * D2 * 2);
  float* NUMS  = (float*)alloc((size_t)SS * D2 * 4);
  float* SKIPS = (float*)alloc((size_t)SS * D2 * 4);
  float* DENS  = (float*)alloc((size_t)SS * 4);
  float* EXS   = (float*)alloc((size_t)ESH * 4);
  float* AGGN  = (float*)alloc((size_t)BS * DD * 4);
  float* JAGG  = (float*)alloc((size_t)BS * 3 * D2 * 4);
  float* LAGG  = (float*)alloc((size_t)BS * 4 * D2 * 4);
  float* SAGG  = (float*)alloc((size_t)BS * 2 * D2 * 4);
  bf* AGGNB = (bf*)alloc((size_t)BS * DD * 2);
  bf* JAGGB = (bf*)alloc((size_t)BS * 3 * D2 * 2);
  bf* LAGGB = (bf*)alloc((size_t)BS * 4 * D2 * 2);
  bf* SAGGB = (bf*)alloc((size_t)BS * 2 * D2 * 2);
  bf* GFB   = (bf*)alloc((size_t)BS * GF * 2);
  float* LATg = (float*)alloc((size_t)BS * 128 * 4);
  float* LATn = (float*)alloc((size_t)BS * 256 * 4);
  float* LATs = (float*)alloc((size_t)BS * 256 * 4);
  float* LATj = (float*)alloc((size_t)BS * 192 * 4);
  float* LATl = (float*)alloc((size_t)BS * 192 * 4);
  // bf16 weights
  bf* aqwB = (bf*)alloc((size_t)DD * DD * 2);
  bf* akwB = (bf*)alloc((size_t)DD * DD * 2);
  bf* avwB = (bf*)alloc((size_t)DD * DD * 2);
  bf* aswB = (bf*)alloc((size_t)DD * DD * 2);
  bf* aewB = (bf*)alloc((size_t)EH * DD * 2);
  bf* sqwB = (bf*)alloc((size_t)D2 * D2 * 2);
  bf* skwB = (bf*)alloc((size_t)D2 * D2 * 2);
  bf* svwB = (bf*)alloc((size_t)D2 * D2 * 2);
  bf* sswB = (bf*)alloc((size_t)D2 * D2 * 2);
  bf* sewB = (bf*)alloc((size_t)3 * D2 * D2 * 2);
  bf* mAtB = (bf*)alloc((size_t)DD * 256 * 2);
  bf* mShB = (bf*)alloc((size_t)2 * D2 * 256 * 2);
  bf* mJoB = (bf*)alloc((size_t)3 * D2 * 192 * 2);
  bf* mLeB = (bf*)alloc((size_t)4 * D2 * 192 * 2);
  bf* mGlB = (bf*)alloc((size_t)GF * 128 * 2);

  // ---- helpers ----
  auto cast = [&](const float* src, bf* dst, size_t n) {
    int blocks = (int)((n + 1023) / 1024);
    if (blocks > 16384) blocks = 16384;
    k_cast_f32_bf16<<<blocks, 256, 0, stream>>>(src, dst, n);
  };
  auto zero = [&](float* p, size_t n) {
    int blocks = (int)((n + 1023) / 1024);
    if (blocks > 16384) blocks = 16384;
    k_zero_f32<<<blocks, 256, 0, stream>>>(p, n);
  };
  auto gemm = [&](const bf* X, const bf* W, const float* bias,
                  float* outF, bf* outB, int M, int K, int Nc, int relu) {
    dim3 g((M + TM - 1) / TM, Nc / TN);
    k_gemm_bf16_wmma<<<g, GEMM_THREADS, 0, stream>>>(X, W, bias, outF, outB, M, K, Nc, relu);
  };

  // ---- cast weights & inputs ----
  cast(node_feats, XB, (size_t)NN * DD);
  cast(graph_features, GFB, (size_t)BS * GF);
  cast(aq_w, aqwB, (size_t)DD * DD);
  cast(ak_w, akwB, (size_t)DD * DD);
  cast(av_w, avwB, (size_t)DD * DD);
  cast(as_w, aswB, (size_t)DD * DD);
  cast(ae_w, aewB, (size_t)EH * DD);
  cast(sq_w, sqwB, (size_t)D2 * D2);
  cast(sk_w, skwB, (size_t)D2 * D2);
  cast(sv_w, svwB, (size_t)D2 * D2);
  cast(ss_w, sswB, (size_t)D2 * D2);
  cast(se_w, sewB, (size_t)3 * D2 * D2);
  cast(m_atoms_w, mAtB, (size_t)DD * 256);
  cast(m_shapes_w, mShB, (size_t)2 * D2 * 256);
  cast(m_joins_w, mJoB, (size_t)3 * D2 * 192);
  cast(m_leafs_w, mLeB, (size_t)4 * D2 * 192);
  cast(m_glob_w, mGlB, (size_t)GF * 128);

  // ---- edge attrs & e = eattr @ we (shared across the 3 conv layers) ----
  k_gather_bonds<<<EE_, 64, 0, stream>>>(bond_table, bond_types, EATTRB, EE_, EH);
  gemm(EATTRB, aewB, nullptr, nullptr, EEB, EE_, EH, DD, 0);

  // ---- 3 atom TransformerConv layers (shared weights) ----
  const float scaleA = 0.0625f;  // 1/sqrt(256)
  bf* hin = XB;
  bf* houts[3] = {HB0, HB1, HB0};
  for (int l = 0; l < 3; ++l) {
    gemm(hin, aqwB, aq_b, nullptr, QB, NN, DD, DD, 0);
    gemm(hin, akwB, ak_b, nullptr, KB, NN, DD, DD, 0);
    gemm(hin, avwB, av_b, nullptr, VB, NN, DD, DD, 0);
    gemm(hin, aswB, as_b, SKIP, nullptr, NN, DD, DD, 0);
    zero(NUM, (size_t)NN * DD);
    zero(DEN, (size_t)NN);
    k_edge_logits<<<(EE_ + 7) / 8, 256, 0, stream>>>(QB, KB, EEB, edge_src, edge_dst,
                                                     EX, DEN, EE_, DD, scaleA);
    k_edge_scatter<<<EE_, 256, 0, stream>>>(VB, EEB, edge_src, edge_dst, EX, NUM, EE_, DD);
    {
      size_t n = (size_t)NN * DD;
      k_combine_relu<<<(unsigned)((n + 255) / 256), 256, 0, stream>>>(NUM, DEN, SKIP,
                                                                      houts[l], NN, DD);
    }
    hin = houts[l];
  }
  bf* HF = houts[2];  // final h (bf16)

  // ---- node-level aggregations ----
  zero(AGGN, (size_t)BS * DD);
  k_accum_rows_bf16<<<NN, 256, 0, stream>>>(HF, node_mol_id, AGGN, NN, DD);

  zero(SHE, (size_t)SS * D2);
  k_accum_ne_seg<<<LNS, 256, 0, stream>>>(XB, HF, nis_atom, nis_shape, SHE, LNS, DD);
  cast(SHE, SHEB, (size_t)SS * D2);

  zero(JAGG, (size_t)BS * 3 * D2);
  k_build_join<<<ESH, 256, 0, stream>>>(SHE, XB, HF, hyper_row, hyper_col, join_idxs,
                                        join_mol_id, JOINB, JAGG, ESH, DD, D2);
  zero(LAGG, (size_t)BS * 4 * D2);
  k_build_leaf<<<NL, 256, 0, stream>>>(SHE, XB, HF, leaf_node_idx, leaf_attach_idx,
                                       leaf_prim_shape, leaf_sec_shape, leaf_mol_id,
                                       LAGG, NL, DD, D2);

  // ---- shape-level TransformerConv x2 (edge attrs = join_emb, fixed) ----
  gemm(JOINB, sewB, nullptr, nullptr, ESB, ESH, 3 * D2, D2, 0);
  const float scaleS = 0.04419417382f;  // 1/sqrt(512)
  bf* sin_ = SHEB;
  bf* souts[2] = {SHB0, SHB1};
  for (int l = 0; l < 2; ++l) {
    gemm(sin_, sqwB, sq_b, nullptr, QSB, SS, D2, D2, 0);
    gemm(sin_, skwB, sk_b, nullptr, KSB, SS, D2, D2, 0);
    gemm(sin_, svwB, sv_b, nullptr, VSB, SS, D2, D2, 0);
    gemm(sin_, sswB, ss_b, SKIPS, nullptr, SS, D2, D2, 0);
    zero(NUMS, (size_t)SS * D2);
    zero(DENS, (size_t)SS);
    k_edge_logits<<<(ESH + 7) / 8, 256, 0, stream>>>(QSB, KSB, ESB, hyper_row, hyper_col,
                                                     EXS, DENS, ESH, D2, scaleS);
    k_edge_scatter<<<ESH, 256, 0, stream>>>(VSB, ESB, hyper_row, hyper_col, EXS, NUMS, ESH, D2);
    {
      size_t n = (size_t)SS * D2;
      k_combine_relu<<<(unsigned)((n + 255) / 256), 256, 0, stream>>>(NUMS, DENS, SKIPS,
                                                                      souts[l], SS, D2);
    }
    sin_ = souts[l];
  }

  zero(SAGG, (size_t)BS * 2 * D2);
  k_accum_shape_cat<<<SS, 256, 0, stream>>>(SHE, SHB1, shape_mol_id, SAGG, SS, D2);

  // ---- final MLPs ----
  cast(AGGN, AGGNB, (size_t)BS * DD);
  cast(JAGG, JAGGB, (size_t)BS * 3 * D2);
  cast(LAGG, LAGGB, (size_t)BS * 4 * D2);
  cast(SAGG, SAGGB, (size_t)BS * 2 * D2);

  gemm(GFB,   mGlB, m_glob_b,   LATg, nullptr, BS, GF,     128, 0);
  gemm(AGGNB, mAtB, m_atoms_b,  LATn, nullptr, BS, DD,     256, 0);
  gemm(SAGGB, mShB, m_shapes_b, LATs, nullptr, BS, 2 * D2, 256, 0);
  gemm(JAGGB, mJoB, m_joins_b,  LATj, nullptr, BS, 3 * D2, 192, 0);
  gemm(LAGGB, mLeB, m_leafs_b,  LATl, nullptr, BS, 4 * D2, 192, 0);

  k_assemble_z<<<BS, 256, 0, stream>>>(LATg, LATn, LATs, LATj, LATl, (float*)d_out, BS);
}
